// MambaBlock_27384711479944
// MI455X (gfx1250) — compile-verified, matrix operand order checked by reading, not run on hardware
//
#include <hip/hip_runtime.h>
#include <hip/hip_bf16.h>

// ---------- types ----------
typedef __attribute__((ext_vector_type(16))) __bf16 v16bf;   // WMMA A/B fragment (32B)
typedef __attribute__((ext_vector_type(8)))  float  v8f;     // WMMA C/D fragment
typedef __attribute__((ext_vector_type(4)))  unsigned int u32x4; // 16B move

struct U32x8 { u32x4 lo, hi; };   // 32B == v16bf

__device__ __forceinline__ unsigned short f2bf(float f) {
    unsigned u = __float_as_uint(f);
    u += 0x7FFFu + ((u >> 16) & 1u);          // round-to-nearest-even
    return (unsigned short)(u >> 16);
}

__device__ __forceinline__ v16bf ld_frag(const unsigned short* p) {
    U32x8 t;
    t.lo = *(const u32x4*)(p);        // K chunk kb..kb+7
    t.hi = *(const u32x4*)(p + 16);   // K chunk kb+16..kb+23
    return __builtin_bit_cast(v16bf, t);
}

// ---------- constants ----------
#define L_SEQ   2048
#define BSZ     2
#define ROWS    (BSZ * L_SEQ)       // 4096
#define DMODEL  512
#define DINNER  2048
#define DSTATE  128
#define NHEADS  32
#define HEADDIM 64
#define CONVDIM 2304
#define NPROJ   4384
#define NPROJP  4480                // padded to 128 multiple
#define LDSW    40                  // LDS row stride (bf16 elems), padded vs 32

// ---------- conversion kernels ----------
__global__ void k_cvt_x(const float* __restrict__ x, unsigned short* __restrict__ xb, int flip) {
    int i = blockIdx.x * 256 + threadIdx.x;            // < ROWS*DMODEL
    int d = i & (DMODEL - 1);
    int row = i >> 9;
    int lf = row & (L_SEQ - 1), b = row >> 11;
    int ls = flip ? (L_SEQ - 1 - lf) : lf;
    xb[i] = f2bf(x[((size_t)(b * L_SEQ + ls)) * DMODEL + d]);
}

__global__ void k_cvt_pad(const float* __restrict__ src, unsigned short* __restrict__ dst,
                          int n, int ntot) {
    int i = blockIdx.x * 256 + threadIdx.x;
    if (i < ntot) dst[i] = (i < n) ? f2bf(src[i]) : (unsigned short)0;
}

// ---------- bf16 TN GEMM with WMMA: C[m,n] = sum_k A[m,k]*B[n,k] ----------
// epi==0: store f32 into C (ldc).  epi==1: out_proj epilogue -> d_out with
// residual add, direction concat (dirbase) and L-flip for the backward pass.
__global__ __launch_bounds__(256) void k_gemm(
    const unsigned short* __restrict__ A, const unsigned short* __restrict__ Bw,
    float* __restrict__ C, int K, int ldc,
    int epi, const float* __restrict__ xres, float* __restrict__ outp,
    int flip, int dirbase)
{
    __shared__ unsigned short As[128 * LDSW];
    __shared__ unsigned short Bs[128 * LDSW];

    const int bm = blockIdx.y << 7, bn = blockIdx.x << 7;
    const int t = threadIdx.x;
    const int lane = t & 31, wave = t >> 5;
    const int wm = (wave >> 2) << 6;      // 0 / 64
    const int wn = (wave & 3) << 5;       // 0,32,64,96
    const int fr = lane & 15;             // fragment row (M or N)
    const int kb = (lane >> 4) << 3;      // K base 0 / 8

    v8f acc[4][2];
#pragma unroll
    for (int i = 0; i < 4; ++i)
#pragma unroll
        for (int j = 0; j < 2; ++j)
            acc[i][j] = (v8f){0.f, 0.f, 0.f, 0.f, 0.f, 0.f, 0.f, 0.f};

    const int lrow = t >> 1;              // 0..127
    const int lcol = (t & 1) << 4;        // 0 / 16
    const unsigned short* ga = A  + (size_t)(bm + lrow) * K + lcol;
    const unsigned short* gb = Bw + (size_t)(bn + lrow) * K + lcol;

    for (int k0 = 0; k0 < K; k0 += 32) {
        u32x4 a0 = *(const u32x4*)(ga + k0);
        u32x4 a1 = *(const u32x4*)(ga + k0 + 8);
        u32x4 b0 = *(const u32x4*)(gb + k0);
        u32x4 b1 = *(const u32x4*)(gb + k0 + 8);
        __syncthreads();                                   // prev-iter reads done
        *(u32x4*)&As[lrow * LDSW + lcol]     = a0;
        *(u32x4*)&As[lrow * LDSW + lcol + 8] = a1;
        *(u32x4*)&Bs[lrow * LDSW + lcol]     = b0;
        *(u32x4*)&Bs[lrow * LDSW + lcol + 8] = b1;
        __syncthreads();

        v16bf af[4], bfz[2];
#pragma unroll
        for (int i = 0; i < 4; ++i)
            af[i] = ld_frag(&As[(wm + (i << 4) + fr) * LDSW + kb]);
#pragma unroll
        for (int j = 0; j < 2; ++j)
            bfz[j] = ld_frag(&Bs[(wn + (j << 4) + fr) * LDSW + kb]);
#pragma unroll
        for (int i = 0; i < 4; ++i)
#pragma unroll
            for (int j = 0; j < 2; ++j)
                acc[i][j] = __builtin_amdgcn_wmma_f32_16x16x32_bf16(
                    false, af[i], false, bfz[j], (short)0, acc[i][j], false, false);
    }

    const int mh = (lane >> 4) << 3;      // D-matrix M offset 0 / 8
#pragma unroll
    for (int i = 0; i < 4; ++i)
#pragma unroll
        for (int j = 0; j < 2; ++j)
#pragma unroll
            for (int r = 0; r < 8; ++r) {
                int gm = bm + wm + (i << 4) + mh + r;
                int gn = bn + wn + (j << 4) + fr;
                float v = acc[i][j][r];
                if (epi == 0) {
                    C[(size_t)gm * ldc + gn] = v;
                } else {
                    int b = gm >> 11, lf = gm & (L_SEQ - 1);
                    int lo = flip ? (L_SEQ - 1 - lf) : lf;
                    size_t orow = ((size_t)b << 11) + lo;
                    outp[orow * (2 * DMODEL) + dirbase + gn] =
                        v + xres[orow * DMODEL + gn];
                }
            }
}

// ---------- depthwise causal conv(4) + SiLU ----------
__global__ void k_conv(const float* __restrict__ zx, const float* __restrict__ cw,
                       const float* __restrict__ cb, float* __restrict__ xc) {
    int i = blockIdx.x * 256 + threadIdx.x;           // < ROWS*CONVDIM
    int c = i % CONVDIM;
    int row = i / CONVDIM;
    int l = row & (L_SEQ - 1), b = row >> 11;
    float s = cb[c];
#pragma unroll
    for (int k = 0; k < 4; ++k) {
        int lk = l - 3 + k;
        if (lk >= 0)
            s += zx[((size_t)(b * L_SEQ + lk)) * NPROJP + DINNER + c] * cw[c * 4 + k];
    }
    xc[(size_t)row * CONVDIM + c] = s / (1.f + __expf(-s));   // SiLU
}

// ---------- dt = softplus(raw + bias); dA = exp(dt * -exp(A_log)) ----------
__global__ void k_dt(const float* __restrict__ zx, const float* __restrict__ dtb,
                     const float* __restrict__ Alog, float* __restrict__ dtv,
                     float* __restrict__ dAv) {
    int i = blockIdx.x * 256 + threadIdx.x;           // < ROWS*NHEADS
    int h = i & (NHEADS - 1);
    int row = i >> 5;
    float raw = zx[(size_t)row * NPROJP + DINNER + CONVDIM + h] + dtb[h];
    float sp = (raw > 20.f) ? raw : log1pf(__expf(raw));
    dtv[i] = sp;
    dAv[i] = __expf(-__expf(Alog[h]) * sp);
}

// ---------- sequential selective scan ----------
// Grid = 64 * ndirs workgroups: blockIdx.x>>6 selects direction (buffers are
// laid out contiguously per direction), so both directions' recurrences run
// concurrently in a single launch when workspace permits.
__global__ __launch_bounds__(256) void k_scan(
    const float* __restrict__ xc, const float* __restrict__ dtv,
    const float* __restrict__ dAv, const float* __restrict__ Dp0,
    const float* __restrict__ Dp1, float* __restrict__ ybuf)
{
    __shared__ float sBC[256];     // B[0..127], C[128..255]
    __shared__ float red[256];
    const int dir = blockIdx.x >> 6;
    const int bh  = blockIdx.x & 63;
    const float* __restrict__ xcd = xc  + (size_t)dir * ROWS * CONVDIM;
    const float* __restrict__ dtd = dtv + (size_t)dir * ROWS * NHEADS;
    const float* __restrict__ dAd = dAv + (size_t)dir * ROWS * NHEADS;
    const float* __restrict__ Dp  = dir ? Dp1 : Dp0;
    float* __restrict__ yd = ybuf + (size_t)dir * ROWS * DINNER;

    int t = threadIdx.x;
    int b = bh >> 5, h = bh & 31;
    int p = t & 63;                // headdim index
    int ng = t >> 6;               // n-quarter 0..3
    int n0 = ng << 5;

    float hreg[32];
#pragma unroll
    for (int i = 0; i < 32; ++i) hreg[i] = 0.f;
    float Dph = Dp[h];

    for (int l = 0; l < L_SEQ; ++l) {
        int row = (b << 11) + l;
        sBC[t] = xcd[(size_t)row * CONVDIM + DINNER + t];     // stage B,C
        if (l + 1 < L_SEQ) {
            // hide next step's B/C + x fetch behind this step's FMA chain
            __builtin_prefetch(xcd + (size_t)(row + 1) * CONVDIM + DINNER + t, 0, 0);
            __builtin_prefetch(xcd + (size_t)(row + 1) * CONVDIM + (h << 6) + p, 0, 0);
        }
        __syncthreads();
        float xv   = xcd[(size_t)row * CONVDIM + (h << 6) + p];
        float dAl  = dAd[(row << 5) + h];
        float coef = dtd[(row << 5) + h] * xv;
        float acc = 0.f;
#pragma unroll
        for (int i = 0; i < 32; ++i) {
            float hv = hreg[i] * dAl + coef * sBC[n0 + i];
            hreg[i] = hv;
            acc += hv * sBC[128 + n0 + i];
        }
        red[t] = acc;
        __syncthreads();
        if (ng == 0) {
            float y = red[p] + red[64 + p] + red[128 + p] + red[192 + p] + Dph * xv;
            yd[(size_t)row * DINNER + (h << 6) + p] = y;
        }
        __syncthreads();
    }
}

// ---------- gate (y * silu(z)) + RMSNorm + bf16 convert ----------
__global__ __launch_bounds__(256) void k_gatenorm(
    const float* __restrict__ ybuf, const float* __restrict__ zx,
    const float* __restrict__ nw, unsigned short* __restrict__ ybb)
{
    __shared__ float red[256];
    int row = blockIdx.x, t = threadIdx.x;
    float v[8];
    float ss = 0.f;
#pragma unroll
    for (int i = 0; i < 8; ++i) {
        int d = t + (i << 8);
        float z = zx[(size_t)row * NPROJP + d];
        float g = ybuf[(size_t)row * DINNER + d] * (z / (1.f + __expf(-z)));
        v[i] = g;
        ss += g * g;
    }
    red[t] = ss;
    __syncthreads();
    for (int s = 128; s > 0; s >>= 1) {
        if (t < s) red[t] += red[t + s];
        __syncthreads();
    }
    float r = rsqrtf(red[0] / (float)DINNER + 1e-5f);
#pragma unroll
    for (int i = 0; i < 8; ++i) {
        int d = t + (i << 8);
        ybb[(size_t)row * DINNER + d] = f2bf(v[i] * r * nw[d]);
    }
}

// ---------- host launcher ----------
extern "C" void kernel_launch(void* const* d_in, const int* in_sizes, int n_in,
                              void* d_out, int out_size, void* d_ws, size_t ws_size,
                              hipStream_t stream) {
    (void)in_sizes; (void)n_in; (void)out_size;
    const float* x = (const float*)d_in[0];
    float* out = (float*)d_out;

    const size_t SZ_ZX  = (size_t)ROWS * NPROJP * 4;
    const size_t SZ_XC  = (size_t)ROWS * CONVDIM * 4;
    const size_t SZ_YB  = (size_t)ROWS * DINNER * 4;
    const size_t SZ_DT  = (size_t)ROWS * NHEADS * 4;
    const size_t SZ_XB  = (size_t)ROWS * DMODEL * 2;
    const size_t SZ_WIN = (size_t)NPROJP * DMODEL * 2;
    const size_t SZ_WO  = (size_t)DMODEL * DINNER * 2;
    const size_t SZ_YBB = (size_t)ROWS * DINNER * 2;

    // Merged-scan mode keeps both directions' scan operands resident so a
    // single 128-WG launch runs both recurrences concurrently (the scan is
    // the latency bottleneck; everything else saturates the chip anyway).
    const size_t need_big = 2 * (SZ_ZX + SZ_XC + SZ_YB + 2 * SZ_DT)
                          + SZ_XB + SZ_WIN + SZ_WO + SZ_YBB + 16 * 256;
    const int ndirs = (ws_size >= need_big) ? 2 : 1;   // ws_size is fixed -> deterministic

    char* ws = (char*)d_ws;
    size_t off = 0;
    auto take = [&](size_t bytes) -> void* {
        void* p = ws + off;
        off += (bytes + 255) & ~(size_t)255;
        return p;
    };
    float* zx   = (float*)take(SZ_ZX * ndirs);
    float* xc   = (float*)take(SZ_XC * ndirs);
    float* ybuf = (float*)take(SZ_YB * ndirs);
    float* dtv  = (float*)take(SZ_DT * ndirs);
    float* dAv  = (float*)take(SZ_DT * ndirs);
    unsigned short* xb   = (unsigned short*)take(SZ_XB);
    unsigned short* winb = (unsigned short*)take(SZ_WIN);
    unsigned short* wout = (unsigned short*)take(SZ_WO);
    unsigned short* ybb  = (unsigned short*)take(SZ_YBB);

    const float* prm[2][8];
    for (int dir = 0; dir < 2; ++dir)
        for (int k = 0; k < 8; ++k)
            prm[dir][k] = (const float*)d_in[1 + dir * 8 + k];
    // prm[dir]: 0=in_w 1=conv_w 2=conv_b 3=dt_bias 4=A_log 5=D 6=norm_w 7=out_w

    for (int pass = 0; pass < (ndirs == 2 ? 1 : 2); ++pass) {
        // ---- phase 1: projections + conv + dt (per direction, buffers per dir in big mode)
        for (int d = 0; d < ndirs; ++d) {
            int dir = (ndirs == 2) ? d : pass;
            float* zxd = zx + (size_t)d * (SZ_ZX / 4);
            float* xcd = xc + (size_t)d * (SZ_XC / 4);
            float* dtd = dtv + (size_t)d * (SZ_DT / 4);
            float* dAd = dAv + (size_t)d * (SZ_DT / 4);

            k_cvt_x<<<(ROWS * DMODEL) / 256, 256, 0, stream>>>(x, xb, dir);
            k_cvt_pad<<<(NPROJP * DMODEL) / 256, 256, 0, stream>>>(
                prm[dir][0], winb, NPROJ * DMODEL, NPROJP * DMODEL);
            k_gemm<<<dim3(NPROJP / 128, ROWS / 128), 256, 0, stream>>>(
                xb, winb, zxd, DMODEL, NPROJP, 0, nullptr, nullptr, 0, 0);
            k_conv<<<(ROWS * CONVDIM) / 256, 256, 0, stream>>>(
                zxd, prm[dir][1], prm[dir][2], xcd);
            k_dt<<<(ROWS * NHEADS) / 256, 256, 0, stream>>>(
                zxd, prm[dir][3], prm[dir][4], dtd, dAd);
        }

        // ---- phase 2: selective scan (both directions in one launch if big mode)
        if (ndirs == 2) {
            k_scan<<<BSZ * NHEADS * 2, 256, 0, stream>>>(
                xc, dtv, dAv, prm[0][5], prm[1][5], ybuf);
        } else {
            k_scan<<<BSZ * NHEADS, 256, 0, stream>>>(
                xc, dtv, dAv, prm[pass][5], prm[pass][5], ybuf);
        }

        // ---- phase 3: gate + RMSNorm + out_proj (+residual, concat, flip)
        for (int d = 0; d < ndirs; ++d) {
            int dir = (ndirs == 2) ? d : pass;
            float* zxd = zx + (size_t)d * (SZ_ZX / 4);
            float* ybd = ybuf + (size_t)d * (SZ_YB / 4);

            k_gatenorm<<<ROWS, 256, 0, stream>>>(ybd, zxd, prm[dir][6], ybb);
            k_cvt_pad<<<(DMODEL * DINNER) / 256, 256, 0, stream>>>(
                prm[dir][7], wout, DMODEL * DINNER, DMODEL * DINNER);
            k_gemm<<<dim3(DMODEL / 128, ROWS / 128), 256, 0, stream>>>(
                ybb, wout, nullptr, DINNER, 0, 1, x, out, dir, dir * DMODEL);
        }
    }
}
